// CustomSigLIP2Attention_82729660055871
// MI455X (gfx1250) — matrix operand order repaired and checked by reading, not compiled, not run on hardware
//
#include <hip/hip_runtime.h>
#include <hip/hip_bf16.h>
#include <stdint.h>

// ---------------------------------------------------------------------------
// Types for CDNA5 WMMA (wave32): v_wmma_f32_16x16x32_bf16
// ---------------------------------------------------------------------------
typedef __bf16 bf16_t;
typedef __attribute__((ext_vector_type(16))) __bf16 v16bf;
typedef __attribute__((ext_vector_type(8)))  __bf16 v8bf;
typedef __attribute__((ext_vector_type(8)))  float  v8f;

union AFrag {
    v16bf v;
    v8bf  h[2];
};

__device__ __forceinline__ v8f wmma_bf16(const v16bf& a, const v16bf& b, const v8f& c) {
    return __builtin_amdgcn_wmma_f32_16x16x32_bf16(false, a, false, b, (short)0, c, false, false);
}

// LDS byte offset of a __shared__ object: flat->LDS truncates to addr[31:0]
__device__ __forceinline__ unsigned lds_off(const void* p) {
    return (unsigned)(size_t)p;
}

// CDNA5 async global->LDS copy, 16B per lane, tracked by ASYNCcnt.
__device__ __forceinline__ void async_copy_b128(unsigned dst_lds, const bf16_t* src) {
    asm volatile("global_load_async_to_lds_b128 %0, %1, off"
                 :: "v"(dst_lds), "v"(src) : "memory");
}
__device__ __forceinline__ void wait_asynccnt0() {
    asm volatile("s_wait_asynccnt 0x0" ::: "memory");
}

// Problem constants
#define Bq   8
#define Sq   1024
#define Dq   1152
#define Hq   16
#define HDq  72
#define HDP  96          // padded head dim (3 x K32 chunks)
#define N3D  3456        // 3*D

// ---------------------------------------------------------------------------
// fp32 -> bf16 conversion
// ---------------------------------------------------------------------------
__global__ void f2bf_kernel(const float* __restrict__ in, bf16_t* __restrict__ out, int n) {
    int i = blockIdx.x * blockDim.x + threadIdx.x;
    if (i < n) out[i] = (bf16_t)in[i];
}

// ---------------------------------------------------------------------------
// zero-fill (32-bit words)
// ---------------------------------------------------------------------------
__global__ void zero_kernel(uint32_t* __restrict__ p, int nwords) {
    int i = blockIdx.x * blockDim.x + threadIdx.x;
    if (i < nwords) p[i] = 0u;
}

// ---------------------------------------------------------------------------
// Tiled WMMA GEMM:  C[M,N] = A[M,K] * W[N,K]^T + bias
//   block tile 128x128, 256 threads = 8 waves (4x2), each wave 32x64.
//   Double-buffered LDS filled by global_load_async_to_lds_b128 (ASYNCcnt),
//   one barrier per K-step; async DMA of tile i+1 overlaps WMMA of tile i.
//   EPI==0: store fp32 to outF
//   EPI==1: QKV epilogue -> scatter bf16 into q[B,H,S,96], k[B,H,S,96],
//           vT[B,H,96,S]
// M % 128 == 0, N % 128 == 0, K % 32 == 0 (true for this problem).
// ---------------------------------------------------------------------------
template <int EPI>
__global__ __launch_bounds__(256) void gemm_bf16_kernel(
    const bf16_t* __restrict__ A, const bf16_t* __restrict__ W,
    const float* __restrict__ bias, float* __restrict__ outF,
    bf16_t* __restrict__ qbuf, bf16_t* __restrict__ kbuf, bf16_t* __restrict__ vT,
    int M, int N, int K)
{
    // 128 rows x 32 cols bf16, padded stride 40; double buffered
    __shared__ bf16_t sA[2][128 * 40];
    __shared__ bf16_t sW[2][128 * 40];

    const int tid  = threadIdx.x;
    const int lane = tid & 31;
    const int wave = tid >> 5;
    const int wm   = wave & 3;        // 0..3 -> row strip of 32
    const int wn   = wave >> 2;       // 0..1 -> col strip of 64
    const int l15  = lane & 15;
    const int hs   = (lane < 16) ? 0 : 1;   // K-half select per WMMA layout

    const int tilesN = N >> 7;
    const int bm = (blockIdx.x / tilesN) << 7;
    const int bn = (blockIdx.x % tilesN) << 7;

    const unsigned ldsA[2] = { lds_off(&sA[0][0]), lds_off(&sA[1][0]) };
    const unsigned ldsW[2] = { lds_off(&sW[0][0]), lds_off(&sW[1][0]) };

    // Tile = 128 rows x 4 chunks of 8 bf16 (16B). 512 chunks, 2 per thread.
    const int c0 = tid, c1 = tid + 256;
    const int r0 = c0 >> 2, o0 = (c0 & 3) << 3;
    const int r1 = c1 >> 2, o1 = (c1 & 3) << 3;

    // issue async loads for K-tile at k0 into buffer `buf`
    auto issue_tile = [&](int k0, int buf) {
        async_copy_b128(ldsA[buf] + (unsigned)(r0 * 40 + o0) * 2,
                        A + (size_t)(bm + r0) * K + k0 + o0);
        async_copy_b128(ldsW[buf] + (unsigned)(r0 * 40 + o0) * 2,
                        W + (size_t)(bn + r0) * K + k0 + o0);
        async_copy_b128(ldsA[buf] + (unsigned)(r1 * 40 + o1) * 2,
                        A + (size_t)(bm + r1) * K + k0 + o1);
        async_copy_b128(ldsW[buf] + (unsigned)(r1 * 40 + o1) * 2,
                        W + (size_t)(bn + r1) * K + k0 + o1);
    };

    v8f acc[2][4] = {};

    const int nt = K >> 5;
    issue_tile(0, 0);

    for (int i = 0; i < nt; ++i) {
        const int cur = i & 1;
        wait_asynccnt0();      // my async stores to LDS for tile i have landed
        __syncthreads();       // everyone's landed; everyone's tile i-1 LDS reads retired
        if (i + 1 < nt)        // overlap DMA of tile i+1 with WMMA of tile i
            issue_tile((i + 1) << 5, cur ^ 1);

        // A fragments: lane<16 holds K{0..15}, lane>=16 K{16..31}
        AFrag af[2], bfr[4];
        #pragma unroll
        for (int mi = 0; mi < 2; ++mi) {
            const bf16_t* p = &sA[cur][(wm * 32 + mi * 16 + l15) * 40 + hs * 16];
            af[mi].h[0] = *(const v8bf*)(p);
            af[mi].h[1] = *(const v8bf*)(p + 8);
        }
        // B fragments: lane holds W-row (== B column), lane<16 K{0..15}, >=16 K{16..31}
        #pragma unroll
        for (int ni = 0; ni < 4; ++ni) {
            const bf16_t* p = &sW[cur][(wn * 64 + ni * 16 + l15) * 40 + hs * 16];
            bfr[ni].h[0] = *(const v8bf*)(p);
            bfr[ni].h[1] = *(const v8bf*)(p + 8);
        }
        #pragma unroll
        for (int mi = 0; mi < 2; ++mi)
            #pragma unroll
            for (int ni = 0; ni < 4; ++ni)
                acc[mi][ni] = wmma_bf16(af[mi].v, bfr[ni].v, acc[mi][ni]);
    }

    // Epilogue. C layout: lane -> col = l15 (+tile), rows = (hs?8:0)+i in vgpr i
    #pragma unroll
    for (int mi = 0; mi < 2; ++mi) {
        #pragma unroll
        for (int ni = 0; ni < 4; ++ni) {
            const int col = bn + wn * 64 + ni * 16 + l15;
            const float bv = bias[col];
            #pragma unroll
            for (int i = 0; i < 8; ++i) {
                const int row = bm + wm * 32 + mi * 16 + hs * 8 + i;
                const float v = acc[mi][ni][i] + bv;
                if (EPI == 0) {
                    outF[(size_t)row * N + col] = v;
                } else {
                    // scatter into q / k / vT (padded head dim)
                    const int which = col / Dq;          // 0=q 1=k 2=v
                    const int r  = col - which * Dq;
                    const int h  = r / HDq;
                    const int hd = r - h * HDq;
                    const int b  = row >> 10;            // /S
                    const int s  = row & (Sq - 1);
                    const size_t bh = (size_t)(b * Hq + h);
                    const bf16_t bw = (bf16_t)v;
                    if (which == 0)      qbuf[(bh * Sq + s) * HDP + hd] = bw;
                    else if (which == 1) kbuf[(bh * Sq + s) * HDP + hd] = bw;
                    else                 vT[(bh * HDP + hd) * Sq + s]   = bw;
                }
            }
        }
    }
}

// ---------------------------------------------------------------------------
// In-place RoPE on q and k: pair (j, j+36) within HD=72, one thread per pair.
// Folds softmax scale (HD^-0.5) into q.
// ---------------------------------------------------------------------------
__global__ void rope_kernel(bf16_t* __restrict__ q, bf16_t* __restrict__ k,
                            const float* __restrict__ cosp, const float* __restrict__ sinp,
                            float qscale, int n)
{
    int idx = blockIdx.x * blockDim.x + threadIdx.x;
    if (idx >= n) return;
    const int j  = idx % 36;
    const int t  = idx / 36;
    const int s  = t & (Sq - 1);
    const int bh = t >> 10;
    const size_t base = ((size_t)bh * Sq + s) * HDP;

    const float c1 = cosp[s * HDq + j],      c2 = cosp[s * HDq + j + 36];
    const float s1 = sinp[s * HDq + j],      s2 = sinp[s * HDq + j + 36];

    float x1 = (float)q[base + j], x2 = (float)q[base + j + 36];
    q[base + j]      = (bf16_t)((x1 * c1 - x2 * s1) * qscale);
    q[base + j + 36] = (bf16_t)((x2 * c2 + x1 * s2) * qscale);

    x1 = (float)k[base + j]; x2 = (float)k[base + j + 36];
    k[base + j]      = (bf16_t)(x1 * c1 - x2 * s1);
    k[base + j + 36] = (bf16_t)(x2 * c2 + x1 * s2);
}

// ---------------------------------------------------------------------------
// Flash attention: one block = 4 waves = 64 q-rows of one (b,h).
// Per wave: Q (16 x 96) resident in 3 A-fragments; loop 32-key tiles:
//   scores (2x3 wmma) -> online softmax -> P transpose via LDS -> PV (6 wmma).
// Output: attn[B*S, 1152] bf16 (normalized by running sum).
// ---------------------------------------------------------------------------
__global__ __launch_bounds__(128) void attn_kernel(
    const bf16_t* __restrict__ qbuf, const bf16_t* __restrict__ kbuf,
    const bf16_t* __restrict__ vT, bf16_t* __restrict__ attnO)
{
    __shared__ bf16_t sP[4][16 * 40];   // per-wave 16x32 P tile, stride 40

    const int lane = threadIdx.x & 31;
    const int wave = threadIdx.x >> 5;
    const int l15  = lane & 15;
    const int hs   = (lane < 16) ? 0 : 1;

    const int qblk = blockIdx.x & 15;        // S/64 = 16 q-blocks
    const int bh   = blockIdx.x >> 4;        // 0..127 (b*16+h)
    const int qrow0 = qblk * 64 + wave * 16;

    const bf16_t* kb = kbuf + (size_t)bh * Sq * HDP;
    const bf16_t* vb = vT   + (size_t)bh * HDP * Sq;

    // Load resident Q fragments (A layout), 3 chunks of K=32
    AFrag qf[3];
    {
        const bf16_t* qrow = qbuf + ((size_t)bh * Sq + qrow0 + l15) * HDP;
        #pragma unroll
        for (int c = 0; c < 3; ++c) {
            const bf16_t* p = qrow + c * 32 + hs * 16;
            qf[c].h[0] = *(const v8bf*)(p);
            qf[c].h[1] = *(const v8bf*)(p + 8);
        }
    }

    v8f O[6] = {};
    float mrun[8], lrun[8];
    #pragma unroll
    for (int i = 0; i < 8; ++i) { mrun[i] = -3.0e38f; lrun[i] = 0.0f; }

    bf16_t* lp = &sP[wave][0];

    for (int kt = 0; kt < Sq; kt += 32) {
        // ---- scores: Q(16x96) x K^T(96x32) -> two 16x16 f32 tiles ----
        v8f s0 = {}, s1 = {};
        #pragma unroll
        for (int c = 0; c < 3; ++c) {
            AFrag b0, b1;
            const bf16_t* p0 = kb + (size_t)(kt + l15) * HDP + c * 32 + hs * 16;
            b0.h[0] = *(const v8bf*)(p0);  b0.h[1] = *(const v8bf*)(p0 + 8);
            const bf16_t* p1 = kb + (size_t)(kt + 16 + l15) * HDP + c * 32 + hs * 16;
            b1.h[0] = *(const v8bf*)(p1);  b1.h[1] = *(const v8bf*)(p1 + 8);
            s0 = wmma_bf16(qf[c].v, b0.v, s0);
            s1 = wmma_bf16(qf[c].v, b1.v, s1);
        }

        // ---- online softmax update (row reductions across 16-lane groups) ----
        float p0f[8], p1f[8];
        #pragma unroll
        for (int i = 0; i < 8; ++i) {
            float mx = fmaxf(s0[i], s1[i]);
            #pragma unroll
            for (int msk = 8; msk > 0; msk >>= 1)
                mx = fmaxf(mx, __shfl_xor(mx, msk, 32));
            const float mnew = fmaxf(mrun[i], mx);
            p0f[i] = __expf(s0[i] - mnew);
            p1f[i] = __expf(s1[i] - mnew);
            float rs = p0f[i] + p1f[i];
            #pragma unroll
            for (int msk = 8; msk > 0; msk >>= 1)
                rs += __shfl_xor(rs, msk, 32);
            const float alpha = __expf(mrun[i] - mnew);
            lrun[i] = lrun[i] * alpha + rs;
            mrun[i] = mnew;
            #pragma unroll
            for (int c = 0; c < 6; ++c) O[c][i] *= alpha;
        }

        // ---- transpose P (C layout -> A layout) through per-wave LDS ----
        #pragma unroll
        for (int i = 0; i < 8; ++i) {
            const int r = hs * 8 + i;
            lp[r * 40 + l15]      = (bf16_t)p0f[i];
            lp[r * 40 + 16 + l15] = (bf16_t)p1f[i];
        }
        asm volatile("s_wait_dscnt 0x0" ::: "memory");   // wave-local LDS RAW
        AFrag pf;
        {
            const bf16_t* p = lp + l15 * 40 + hs * 16;
            pf.h[0] = *(const v8bf*)(p);
            pf.h[1] = *(const v8bf*)(p + 8);
        }

        // ---- O += P(16x32) x V(32x96) : 6 column chunks ----
        #pragma unroll
        for (int c = 0; c < 6; ++c) {
            AFrag vf;
            const bf16_t* p = vb + (size_t)(c * 16 + l15) * Sq + kt + hs * 16;
            vf.h[0] = *(const v8bf*)(p);  vf.h[1] = *(const v8bf*)(p + 8);
            O[c] = wmma_bf16(pf.v, vf.v, O[c]);
        }
    }

    // ---- epilogue: normalize and store bf16 into attn[B*S, 1152] ----
    const int b = bh >> 4, h = bh & 15;
    #pragma unroll
    for (int c = 0; c < 6; ++c) {
        const int hd = c * 16 + l15;
        if (hd < HDq) {
            #pragma unroll
            for (int i = 0; i < 8; ++i) {
                const int srow = qrow0 + hs * 8 + i;
                const float v = O[c][i] * (1.0f / lrun[i]);
                attnO[(size_t)(b * Sq + srow) * Dq + h * HDq + hd] = (bf16_t)v;
            }
        }
    }
}

// ---------------------------------------------------------------------------
// Host launch
// ---------------------------------------------------------------------------
extern "C" void kernel_launch(void* const* d_in, const int* in_sizes, int n_in,
                              void* d_out, int out_size, void* d_ws, size_t ws_size,
                              hipStream_t stream) {
    (void)in_sizes; (void)n_in; (void)out_size; (void)ws_size;

    const float* hidden = (const float*)d_in[0];
    const float* cosp   = (const float*)d_in[1];
    const float* sinp   = (const float*)d_in[2];
    const float* qkv_w  = (const float*)d_in[3];
    const float* qkv_b  = (const float*)d_in[4];
    const float* proj_w = (const float*)d_in[5];
    const float* proj_b = (const float*)d_in[6];
    float* out = (float*)d_out;

    // workspace carve-up (all sizes multiples of 256B)
    char* ws = (char*)d_ws;
    size_t off = 0;
    bf16_t* hB    = (bf16_t*)(ws + off); off += (size_t)Bq * Sq * Dq * 2;        // 18.9MB
    bf16_t* wqkvB = (bf16_t*)(ws + off); off += (size_t)N3D * Dq * 2;            // 8.0MB
    bf16_t* wprjB = (bf16_t*)(ws + off); off += (size_t)Dq * Dq * 2;             // 2.65MB
    bf16_t* qbuf  = (bf16_t*)(ws + off); off += (size_t)Bq * Hq * Sq * HDP * 2;  // 25.2MB
    bf16_t* kbuf  = (bf16_t*)(ws + off); off += (size_t)Bq * Hq * Sq * HDP * 2;
    bf16_t* vT    = (bf16_t*)(ws + off); off += (size_t)Bq * Hq * HDP * Sq * 2;
    bf16_t* attnB = (bf16_t*)(ws + off); off += (size_t)Bq * Sq * Dq * 2;

    const int nh  = Bq * Sq * Dq;        // 9437184
    const int nw1 = N3D * Dq;            // 3981312
    const int nw2 = Dq * Dq;             // 1327104

    // 1) convert inputs to bf16
    f2bf_kernel<<<(nh  + 255) / 256, 256, 0, stream>>>(hidden, hB,    nh);
    f2bf_kernel<<<(nw1 + 255) / 256, 256, 0, stream>>>(qkv_w,  wqkvB, nw1);
    f2bf_kernel<<<(nw2 + 255) / 256, 256, 0, stream>>>(proj_w, wprjB, nw2);

    // 2) zero padded q/k/vT region (contiguous: 3 * 25165824 bytes)
    const int zwords = (3 * Bq * Hq * Sq * HDP * 2) / 4;  // 18874368
    zero_kernel<<<(zwords + 255) / 256, 256, 0, stream>>>((uint32_t*)qbuf, zwords);

    // 3) QKV GEMM with scatter epilogue: M=8192, N=3456, K=1152
    {
        const int M = Bq * Sq, N = N3D, K = Dq;
        dim3 grid((M / 128) * (N / 128));   // 1728
        gemm_bf16_kernel<1><<<grid, 256, 0, stream>>>(hB, wqkvB, qkv_b, nullptr,
                                                      qbuf, kbuf, vT, M, N, K);
    }

    // 4) RoPE in place on q,k (scale folded into q)
    {
        const int n = Bq * Hq * Sq * 36;  // 4718592 pairs
        const float qscale = 0.11785113019775792f;  // 72^-0.5
        rope_kernel<<<(n + 255) / 256, 256, 0, stream>>>(qbuf, kbuf, cosp, sinp, qscale, n);
    }

    // 5) flash attention: 2048 blocks x 128 threads
    attn_kernel<<<Bq * Hq * (Sq / 64), 128, 0, stream>>>(qbuf, kbuf, vT, attnB);

    // 6) output projection: M=8192, N=1152, K=1152 -> fp32 d_out
    {
        const int M = Bq * Sq, N = Dq, K = Dq;
        dim3 grid((M / 128) * (N / 128));   // 576
        gemm_bf16_kernel<0><<<grid, 256, 0, stream>>>(attnB, wprjB, proj_b, out,
                                                      nullptr, nullptr, nullptr, M, N, K);
    }
}
